// GCNGraphClassifier_3_13606456394539
// MI455X (gfx1250) — matrix operand order repaired and checked by reading, not compiled, not run on hardware
//
#include <hip/hip_runtime.h>

// ---------------------------------------------------------------------------
// GCN graph classifier for MI455X (gfx1250, wave32).
//   - activations kept in fp32 (scatter accumulator) + packed bf16 (GEMM input)
//   - 3x { H = Xbf @ Wbf (v_wmma_f32_16x16x32_bf16, f32 acc, W staged in LDS
//          via global_load_async_to_lds_b128);  float4 scatter-add norm*H[src];
//          fused bias+relu writing fp32 + bf16 copies (self-loops folded) }
//   - LDS-accumulated mean pool over 64 graphs, tiny final linear.
// ---------------------------------------------------------------------------

typedef __attribute__((ext_vector_type(16))) __bf16 v16bf;
typedef __attribute__((ext_vector_type(8)))  float  v8f;
typedef __attribute__((ext_vector_type(4)))  int    v4i;

#define NN   50000
#define EE   600000
#define HD   128
#define OUTD 10
#define NG   64

#if __has_builtin(__builtin_amdgcn_global_load_async_to_lds_b128) && \
    __has_builtin(__builtin_amdgcn_s_wait_asynccnt)
#define HAVE_ASYNC_LDS 1
#endif

#define ASG __attribute__((address_space(1)))
#define ASL __attribute__((address_space(3)))

__device__ __forceinline__ unsigned short f2bf(float f) {
    // round-to-nearest-even fp32 -> bf16
    unsigned int u = __float_as_uint(f);
    u = (u + 0x7FFFu + ((u >> 16) & 1u)) >> 16;
    return (unsigned short)u;
}

__device__ __forceinline__ uint2 pack4bf(float4 f) {
    uint2 r;
    r.x = (unsigned)f2bf(f.x) | ((unsigned)f2bf(f.y) << 16);
    r.y = (unsigned)f2bf(f.z) | ((unsigned)f2bf(f.w) << 16);
    return r;
}

union V16U {
    uint4 q[2];
    v16bf v;
};

// ---------------------------------------------------------------- utilities
__global__ void k_fill_zero4(float4* __restrict__ p, int n4) {
    int i = blockIdx.x * blockDim.x + threadIdx.x;
    const float4 z = {0.f, 0.f, 0.f, 0.f};
    if (i < n4) p[i] = z;
}

__global__ void k_init_deg(float* __restrict__ deg, int n) {
    int i = blockIdx.x * blockDim.x + threadIdx.x;
    if (i < n) deg[i] = 1.0f;                       // self loop contributes 1
}

__global__ void k_deg_edges(const long long* __restrict__ ei,
                            float* __restrict__ deg, int E) {
    int e = blockIdx.x * blockDim.x + threadIdx.x;
    if (e < E) atomicAdd(&deg[(int)ei[(size_t)E + e]], 1.0f);
}

__global__ void k_rsqrt_inplace(float* __restrict__ d, int n) {
    int i = blockIdx.x * blockDim.x + threadIdx.x;
    if (i < n) d[i] = rsqrtf(d[i]);                 // deg >= 1 always
}

__global__ void k_edge_norm(const long long* __restrict__ ei,
                            const float* __restrict__ dinv,
                            float* __restrict__ nrm, int E) {
    int e = blockIdx.x * blockDim.x + threadIdx.x;
    if (e < E) nrm[e] = dinv[(int)ei[e]] * dinv[(int)ei[(size_t)E + e]];
}

__global__ void k_w2bf(const float* __restrict__ W,
                       unsigned short* __restrict__ o, int n) {
    int i = blockIdx.x * blockDim.x + threadIdx.x;
    if (i < n) o[i] = f2bf(W[i]);
}

// x (fp32) -> packed bf16, 4 elements / thread
__global__ void k_x2bf(const float4* __restrict__ x, uint2* __restrict__ xb, int n4) {
    int i = blockIdx.x * blockDim.x + threadIdx.x;
    if (i < n4) xb[i] = pack4bf(x[i]);
}

// ------------------------------------------------------------- WMMA GEMM
// C[N,128] = A[N,128](bf16) @ W(128x128 bf16, staged in LDS).  One wave per
// 16-row strip; 8 accumulator tiles; K loop 4 x 32 (v_wmma_f32_16x16x32_bf16).
__global__ void __launch_bounds__(256)
k_gemm_wmma(const unsigned short* __restrict__ Abf,
            const unsigned short* __restrict__ Wb,
            float* __restrict__ C, int nStrips) {
    __shared__ unsigned short sW[HD * HD];          // 32 KB bf16 weights

    const int tid = threadIdx.x;
    {
        const uint4* gw = (const uint4*)Wb;
        uint4*       lw = (uint4*)sW;
#ifdef HAVE_ASYNC_LDS
        // CDNA5 async DMA: 2048 x b128 straight into LDS, no VGPR round trip
#pragma unroll
        for (int i = 0; i < 8; ++i) {
            __builtin_amdgcn_global_load_async_to_lds_b128(
                (ASG v4i*)(gw + tid + 256 * i),
                (ASL v4i*)(lw + tid + 256 * i), 0, 0);
        }
        __builtin_amdgcn_s_wait_asynccnt(0);
#else
#pragma unroll
        for (int i = 0; i < 8; ++i) lw[tid + 256 * i] = gw[tid + 256 * i];
#endif
    }
    __syncthreads();

    const int wave  = tid >> 5;
    const int lane  = tid & 31;
    const int strip = blockIdx.x * 8 + wave;
    if (strip >= nStrips) return;

    const int m0    = strip * 16;
    const int row   = m0 + (lane & 15);
    const int khalf = (lane >> 4) << 3;             // 0 or 8

    v8f acc[8];
    const v8f vzero = {0.f, 0.f, 0.f, 0.f, 0.f, 0.f, 0.f, 0.f};
#pragma unroll
    for (int n = 0; n < 8; ++n) acc[n] = vzero;

#pragma unroll
    for (int ks = 0; ks < 4; ++ks) {
        const int k0 = ks * 32;
        // A operand (CDNA5 16-bit A layout): K = kb..kb+7 and kb+16..kb+23,
        // 8 consecutive bf16 = one b128 load each.
        const uint4* ap = (const uint4*)(Abf + (size_t)row * HD + k0 + khalf);
        V16U a;
        a.q[0] = ap[0];
        a.q[1] = ap[2];

        // B operand: lane%16 = N column, K in same order as A (kb..kb+7, kb+16..kb+23) from LDS
        const unsigned short* bp = sW + (k0 + khalf) * HD + (lane & 15);
#pragma unroll
        for (int n = 0; n < 8; ++n) {
            V16U b;
            unsigned short* bs = (unsigned short*)&b;
#pragma unroll
            for (int s = 0; s < 16; ++s)
                bs[s] = bp[(s < 8 ? s : s + 8) * HD + n * 16];
            acc[n] = __builtin_amdgcn_wmma_f32_16x16x32_bf16(
                false, a.v, false, b.v, (short)0, acc[n], false, false);
        }
    }

    // C/D layout: VGPR r -> row m0 + r + 8*(lane>=16); col = lane&15
    const int rbase = m0 + ((lane >> 4) << 3);
    const int col   = lane & 15;
#pragma unroll
    for (int n = 0; n < 8; ++n) {
#pragma unroll
        for (int r = 0; r < 8; ++r) {
            C[(size_t)(rbase + r) * HD + n * 16 + col] = acc[n][r];
        }
    }
}

// --------------------------------------------------- edge scatter (dominant)
// one thread per (edge, 4 features): float4 gather, 4 global atomic f32 adds
__global__ void k_aggregate4(const long long* __restrict__ ei,
                             const float* __restrict__ nrm,
                             const float4* __restrict__ H4,
                             float* __restrict__ out, int E) {
    int gid = blockIdx.x * blockDim.x + threadIdx.x;
    if (gid >= E * 32) return;
    int e = gid >> 5;
    int q = gid & 31;
    int s = (int)ei[e];
    int d = (int)ei[(size_t)E + e];
    float  nr = nrm[e];
    float4 h  = H4[(size_t)s * 32 + q];
    float* o  = out + (size_t)d * HD + q * 4;
    atomicAdd(o + 0, nr * h.x);
    atomicAdd(o + 1, nr * h.y);
    atomicAdd(o + 2, nr * h.z);
    atomicAdd(o + 3, nr * h.w);
}

// self-loop term (dinv^2 * Hlin) + bias + relu; writes fp32 (accumulator /
// pooling input) and packed bf16 (next layer's GEMM input)
__global__ void k_bias_relu4(float4* __restrict__ Hact4,
                             const float4* __restrict__ Hlin4,
                             const float* __restrict__ dinv,
                             const float4* __restrict__ b4,
                             uint2* __restrict__ Hbf, int n4) {
    int gid = blockIdx.x * blockDim.x + threadIdx.x;
    if (gid >= n4) return;
    int i  = gid >> 5;
    int q  = gid & 31;
    float  di = dinv[i];
    float  d2 = di * di;
    float4 a  = Hact4[gid];
    float4 h  = Hlin4[gid];
    float4 bb = b4[q];
    float4 v;
    v.x = fmaxf(a.x + d2 * h.x + bb.x, 0.0f);
    v.y = fmaxf(a.y + d2 * h.y + bb.y, 0.0f);
    v.z = fmaxf(a.z + d2 * h.z + bb.z, 0.0f);
    v.w = fmaxf(a.w + d2 * h.w + bb.w, 0.0f);
    Hact4[gid] = v;
    Hbf[gid]   = pack4bf(v);
}

// ----------------------------------------------------- mean pool over graphs
__global__ void __launch_bounds__(256)
k_pool(const float* __restrict__ H, const long long* __restrict__ batch,
       float* __restrict__ sums, float* __restrict__ cnts, int n, int perBlock) {
    __shared__ float ls[NG * HD];                   // 32 KB partial sums
    __shared__ float lc[NG];
    int tid = threadIdx.x;
    for (int i = tid; i < NG * HD; i += 256) ls[i] = 0.0f;
    if (tid < NG) lc[tid] = 0.0f;
    __syncthreads();

    int start = blockIdx.x * perBlock;
    int end   = start + perBlock;
    if (end > n) end = n;
    int f = tid & 127;
    for (int base = start; base < end; base += 2) {
        int node = base + (tid >> 7);
        if (node < end) {
            int g = (int)batch[node];
            atomicAdd(&ls[g * HD + f], H[(size_t)node * HD + f]);  // ds_add_f32
            if (f == 0) atomicAdd(&lc[g], 1.0f);
        }
    }
    __syncthreads();
    for (int i = tid; i < NG * HD; i += 256) atomicAdd(&sums[i], ls[i]);
    if (tid < NG) atomicAdd(&cnts[tid], lc[tid]);
}

// --------------------------------------------------------- final tiny linear
__global__ void k_final(const float* __restrict__ sums,
                        const float* __restrict__ cnts,
                        const float* __restrict__ Wl,
                        const float* __restrict__ bl,
                        float* __restrict__ out) {
    int t = blockIdx.x * blockDim.x + threadIdx.x;
    if (t >= NG * OUTD) return;
    int g = t / OUTD, o = t % OUTD;
    float c   = fmaxf(cnts[g], 1.0f);
    float acc = bl[o];
#pragma unroll 4
    for (int k = 0; k < HD; ++k)
        acc += (sums[g * HD + k] / c) * Wl[k * OUTD + o];
    out[t] = acc;
}

// ---------------------------------------------------------------------------
extern "C" void kernel_launch(void* const* d_in, const int* in_sizes, int n_in,
                              void* d_out, int out_size, void* d_ws, size_t ws_size,
                              hipStream_t stream) {
    const float*     x     = (const float*)d_in[0];
    const long long* ei    = (const long long*)d_in[1];   // [2, E] int64
    const long long* batch = (const long long*)d_in[2];   // [N] int64
    const float* Wmat[3]   = {(const float*)d_in[3], (const float*)d_in[5],
                              (const float*)d_in[7]};
    const float* bias[3]   = {(const float*)d_in[4], (const float*)d_in[6],
                              (const float*)d_in[8]};
    const float* Wlin      = (const float*)d_in[9];
    const float* blin      = (const float*)d_in[10];
    float*       out       = (float*)d_out;

    // workspace carve-up (256B aligned)
    char* p = (char*)d_ws;
    auto alloc = [&](size_t bytes) -> char* {
        char* r = p;
        p += (bytes + 255) & ~(size_t)255;
        return r;
    };
    float*          dinv = (float*)alloc((size_t)NN * 4);
    float*          nrm  = (float*)alloc((size_t)EE * 4);
    unsigned short* Wbf  = (unsigned short*)alloc((size_t)3 * HD * HD * 2);
    float*          Hlin = (float*)alloc((size_t)NN * HD * 4);
    float*          Hact = (float*)alloc((size_t)NN * HD * 4);
    unsigned short* Hbf  = (unsigned short*)alloc((size_t)NN * HD * 2);
    float*          sums = (float*)alloc((size_t)(NG * HD + NG) * 4);
    float*          cnts = sums + NG * HD;

    dim3 B(256);
    // degree / normalization
    k_init_deg<<<(NN + 255) / 256, B, 0, stream>>>(dinv, NN);
    k_deg_edges<<<(EE + 255) / 256, B, 0, stream>>>(ei, dinv, EE);
    k_rsqrt_inplace<<<(NN + 255) / 256, B, 0, stream>>>(dinv, NN);
    k_edge_norm<<<(EE + 255) / 256, B, 0, stream>>>(ei, dinv, nrm, EE);
    for (int l = 0; l < 3; ++l)
        k_w2bf<<<(HD * HD + 255) / 256, B, 0, stream>>>(Wmat[l],
                                                        Wbf + l * HD * HD, HD * HD);

    const int nStrips = NN / 16;      // 3125 (N divisible by 16)
    const int N4      = NN * 32;      // N*HD/4 float4 groups
    // initial activations -> bf16
    k_x2bf<<<(N4 + 255) / 256, B, 0, stream>>>((const float4*)x, (uint2*)Hbf, N4);

    for (int l = 0; l < 3; ++l) {
        k_gemm_wmma<<<(nStrips + 7) / 8, B, 0, stream>>>(
            Hbf, Wbf + l * HD * HD, Hlin, nStrips);
        k_fill_zero4<<<(N4 + 255) / 256, B, 0, stream>>>((float4*)Hact, N4);
        k_aggregate4<<<(EE * 32 + 255) / 256, B, 0, stream>>>(
            ei, nrm, (const float4*)Hlin, Hact, EE);
        k_bias_relu4<<<(N4 + 255) / 256, B, 0, stream>>>(
            (float4*)Hact, (const float4*)Hlin, dinv,
            (const float4*)bias[l], (uint2*)Hbf, N4);
    }

    k_fill_zero4<<<((NG * HD + NG) / 4 + 255) / 256, B, 0, stream>>>(
        (float4*)sums, (NG * HD + NG) / 4);
    int perBlock = (NN + 127) / 128;   // 391 nodes per pooling block
    k_pool<<<128, B, 0, stream>>>(Hact, batch, sums, cnts, NN, perBlock);
    k_final<<<(NG * OUTD + 255) / 256, B, 0, stream>>>(sums, cnts, Wlin, blin, out);
}
